// MultiHeadAttentionLayer_attn_2972117369102
// MI455X (gfx1250) — matrix operand order repaired and checked by reading, not compiled, not run on hardware
//
#include <hip/hip_runtime.h>
#include <hip/hip_bf16.h>

typedef __bf16 bf16_t;
typedef __attribute__((ext_vector_type(16))) __bf16 v16bf;
typedef __attribute__((ext_vector_type(8)))  __bf16 v8bf;
typedef __attribute__((ext_vector_type(8)))  float  v8f;

#define B_DIM 4
#define S_DIM 2048
#define D_MODEL 1024
#define NHEAD 16
#define HDIM 64

__device__ __forceinline__ v16bf cat8(v8bf lo, v8bf hi) {
  return __builtin_shufflevector(lo, hi, 0,1,2,3,4,5,6,7,8,9,10,11,12,13,14,15);
}
__device__ __forceinline__ v8f wmma_bf16(v16bf a, v16bf b, v8f c) {
  // D = A(16x32 bf16) * B(32x16 bf16) + C(16x16 f32)
  return __builtin_amdgcn_wmma_f32_16x16x32_bf16(false, a, false, b, (short)0, c, false, false);
}
__device__ __forceinline__ v8f zero8() {
  v8f z = {0.f,0.f,0.f,0.f,0.f,0.f,0.f,0.f};
  return z;
}

// Async DMA: global -> LDS, 16B per lane, tracked by ASYNCcnt (cdna5 §08).
// LDS byte address = low 32 bits of the generic pointer (flat-LDS aperture).
__device__ __forceinline__ void async_copy_b128(void* lds_dst, const void* gsrc) {
  unsigned lds_addr = (unsigned)(uintptr_t)lds_dst;
  unsigned long long gaddr = (unsigned long long)(uintptr_t)gsrc;
  asm volatile("global_load_async_to_lds_b128 %0, %1, off"
               :: "v"(lds_addr), "v"(gaddr)
               : "memory");
}
__device__ __forceinline__ void wait_async0() {
  asm volatile("s_wait_asynccnt 0x0" ::: "memory");
}

// ---------------------------------------------------------------------------
// Stage 1: weight f32 [k][n] -> bf16 transposed [n][k] (B-fragments contiguous)
// ---------------------------------------------------------------------------
__global__ __launch_bounds__(256)
void wt_bf16_kernel(const float* __restrict__ W, bf16_t* __restrict__ Wt) {
  int i = blockIdx.x * 256 + threadIdx.x;      // 0 .. 1M-1
  int n = i & (D_MODEL - 1);
  int k = i >> 10;
  Wt[(size_t)n * D_MODEL + k] = (bf16_t)W[(size_t)k * D_MODEL + n];
}

// ---------------------------------------------------------------------------
// Stage 2: projection GEMM  Y = x @ W + b
// Block = 4 waves x 16 rows, all sharing one 64-col slice (= head h).
// The 64x32 bf16 weight chunk is async-DMA'd into double-buffered LDS once
// per block (4x traffic cut vs per-wave loads); chunk kc+1 prefetched while
// WMMAs run on kc. vmode=0: store [B,H,S,D]; vmode=1 (V): store [B,H,D,S].
// ---------------------------------------------------------------------------
__global__ __launch_bounds__(128)
void proj_gemm_kernel(const float* __restrict__ x, const bf16_t* __restrict__ Wt,
                      const float* __restrict__ bias, bf16_t* __restrict__ out,
                      int vmode) {
  __shared__ alignas(16) bf16_t Wtile[2][64][40];   // [buf][n-local][k], pad 32->40

  const int tid = threadIdx.x, w = tid >> 5, lane = tid & 31;
  const int l = lane & 15, half = lane >> 4;
  const int rowblk = (blockIdx.x * 4 + w) * 16;       // global token row
  const int h = blockIdx.y;
  const int b = rowblk >> 11;
  const int sloc = rowblk & (S_DIM - 1);

  const float* xrow = x + (size_t)(rowblk + l) * D_MODEL;

  // Staging coords: 128 threads x 16B = 2KB per chunk (64 n-rows x k 0-7,16-23)
  const int wn = tid >> 1, wseg = tid & 1;
  const bf16_t* wsrc = Wt + (size_t)(h * HDIM + wn) * D_MODEL + wseg * 16;
  async_copy_b128(&Wtile[0][wn][wseg * 16], wsrc);

  v8f acc[4];
  #pragma unroll
  for (int c = 0; c < 4; ++c) acc[c] = zero8();

  #pragma unroll 2
  for (int kc = 0; kc < 32; ++kc) {
    const int cur = kc & 1;
    // Issue A-fragment global loads early (independent of the LDS wait)
    const int c0 = kc * 32 + half * 8;
    float4 f0 = *(const float4*)(xrow + c0);
    float4 f1 = *(const float4*)(xrow + c0 + 4);
    float4 f2 = *(const float4*)(xrow + c0 + 16);
    float4 f3 = *(const float4*)(xrow + c0 + 20);

    wait_async0();        // my DMA into buf `cur` done
    __syncthreads();      // everyone's done; prev compute on buf cur^1 done
    if (kc + 1 < 32)      // prefetch next weight chunk into the other buffer
      async_copy_b128(&Wtile[cur ^ 1][wn][wseg * 16], wsrc + (kc + 1) * 32);

    v16bf a;
    a[0]=(bf16_t)f0.x; a[1]=(bf16_t)f0.y; a[2]=(bf16_t)f0.z; a[3]=(bf16_t)f0.w;
    a[4]=(bf16_t)f1.x; a[5]=(bf16_t)f1.y; a[6]=(bf16_t)f1.z; a[7]=(bf16_t)f1.w;
    a[8]=(bf16_t)f2.x; a[9]=(bf16_t)f2.y; a[10]=(bf16_t)f2.z; a[11]=(bf16_t)f2.w;
    a[12]=(bf16_t)f3.x; a[13]=(bf16_t)f3.y; a[14]=(bf16_t)f3.z; a[15]=(bf16_t)f3.w;

    #pragma unroll
    for (int c = 0; c < 4; ++c) {
      const bf16_t* wp = &Wtile[cur][c * 16 + l][half * 8];
      v16bf bb = cat8(*(const v8bf*)wp, *(const v8bf*)(wp + 16));
      acc[c] = wmma_bf16(a, bb, acc[c]);
    }
    __syncthreads();      // reads of buf `cur` done before it is overwritten
  }

  if (vmode) {  // V: [B,H,D,S], pack 8 consecutive s per lane -> one b128 store
    size_t vbase = (size_t)(b * NHEAD + h) * HDIM * S_DIM;
    #pragma unroll
    for (int c = 0; c < 4; ++c) {
      float bv = bias[h * HDIM + c * 16 + l];
      v8bf pk;
      #pragma unroll
      for (int r = 0; r < 8; ++r) pk[r] = (bf16_t)(acc[c][r] + bv);
      *(v8bf*)(out + vbase + (size_t)(c * 16 + l) * S_DIM + sloc + half * 8) = pk;
    }
  } else {      // Q/K: [B,H,S,D]
    size_t qbase = ((size_t)(b * NHEAD + h) * S_DIM + sloc) * HDIM;
    #pragma unroll
    for (int c = 0; c < 4; ++c) {
      float bv = bias[h * HDIM + c * 16 + l];
      #pragma unroll
      for (int r = 0; r < 8; ++r) {
        int M = r + half * 8;  // C layout: VGPR r, lane half selects +8
        out[qbase + (size_t)M * HDIM + c * 16 + l] = (bf16_t)(acc[c][r] + bv);
      }
    }
  }
}

// ---------------------------------------------------------------------------
// Stage 3: flash attention. Block = 4 waves = 64 query rows of one (b,h).
// K/V chunks (32 keys) DMA'd into double-buffered LDS via async-to-LDS ops,
// prefetching chunk tc+1 while WMMAs run on chunk tc. Online softmax in VGPRs;
// P relaid C-layout -> A-layout via a 1KB per-wave LDS tile.
// ---------------------------------------------------------------------------
__global__ __launch_bounds__(128)
void attn_kernel(const bf16_t* __restrict__ Qb, const bf16_t* __restrict__ Kb,
                 const bf16_t* __restrict__ Vt, bf16_t* __restrict__ Ob) {
  __shared__ alignas(16) bf16_t Ks[2][32][72];    // [buf][t-local][d], pad 64->72
  __shared__ alignas(16) bf16_t Vs[2][64][40];    // [buf][d][t-local], pad 32->40
  __shared__ alignas(16) bf16_t Pw[4][16][40];    // per-wave P tile, pad 32->40

  const int tid = threadIdx.x, w = tid >> 5, lane = tid & 31;
  const int l = lane & 15, half = lane >> 4;
  const int h = blockIdx.y, b = blockIdx.z;
  const int bh = b * NHEAD + h;
  const int q0 = blockIdx.x * 64 + w * 16;

  const bf16_t* Qh = Qb + (size_t)bh * S_DIM * HDIM;
  const bf16_t* Kh = Kb + (size_t)bh * S_DIM * HDIM;
  const bf16_t* Vh = Vt + (size_t)bh * HDIM * S_DIM;

  // Per-thread staging coordinates (16B of K and 16B of V per thread per chunk)
  const int krow = tid >> 2, kseg = tid & 3;
  const int vd = tid >> 1, vseg = tid & 1;

  // Q A-fragments (d = 0..31 and 32..63), row = q0 + l
  const bf16_t* qp = Qh + (size_t)(q0 + l) * HDIM;
  const int c0 = half * 16;
  v16bf qa0 = cat8(*(const v8bf*)(qp + c0),      *(const v8bf*)(qp + c0 + 8));
  v16bf qa1 = cat8(*(const v8bf*)(qp + 32 + c0), *(const v8bf*)(qp + 32 + c0 + 8));

  v8f o[4];
  #pragma unroll
  for (int c = 0; c < 4; ++c) o[c] = zero8();
  float m[8], lsum[8];
  #pragma unroll
  for (int r = 0; r < 8; ++r) { m[r] = -1e30f; lsum[r] = 0.f; }

  // Prologue: async-stage chunk 0 into buffer 0
  async_copy_b128(&Ks[0][krow][kseg * 16], Kh + (size_t)krow * HDIM + kseg * 16);
  async_copy_b128(&Vs[0][vd][vseg * 16],   Vh + (size_t)vd * S_DIM + vseg * 16);

  const int NT = S_DIM / 32;
  for (int tc = 0; tc < NT; ++tc) {
    const int cur = tc & 1;
    wait_async0();       // my async copies into buf `cur` are done
    __syncthreads();     // everyone's are done; prev compute on buf cur^1 done

    // Prefetch next chunk into the other buffer while we compute this one
    if (tc + 1 < NT) {
      const int t1 = (tc + 1) * 32;
      async_copy_b128(&Ks[cur ^ 1][krow][kseg * 16],
                      Kh + (size_t)(t1 + krow) * HDIM + kseg * 16);
      async_copy_b128(&Vs[cur ^ 1][vd][vseg * 16],
                      Vh + (size_t)vd * S_DIM + t1 + vseg * 16);
    }

    // Scores: S[16 x 32] = Q(16x64) . K^T; 4 WMMAs
    const int kd = half * 16;
    v16bf ka_lo = cat8(*(const v8bf*)&Ks[cur][l][kd],      *(const v8bf*)&Ks[cur][l][kd + 8]);
    v16bf ka_hi = cat8(*(const v8bf*)&Ks[cur][l][32 + kd], *(const v8bf*)&Ks[cur][l][32 + kd + 8]);
    v16bf kb_lo = cat8(*(const v8bf*)&Ks[cur][16 + l][kd],      *(const v8bf*)&Ks[cur][16 + l][kd + 8]);
    v16bf kb_hi = cat8(*(const v8bf*)&Ks[cur][16 + l][32 + kd], *(const v8bf*)&Ks[cur][16 + l][32 + kd + 8]);
    v8f Sa = zero8(), Sb = zero8();
    Sa = wmma_bf16(qa0, ka_lo, Sa);  Sa = wmma_bf16(qa1, ka_hi, Sa);
    Sb = wmma_bf16(qa0, kb_lo, Sb);  Sb = wmma_bf16(qa1, kb_hi, Sb);

    // Online softmax over the 32-wide chunk. Row M = r + 8*half lives in the
    // 16-lane half-wave, so shfl_xor masks {1,2,4,8} give exact row reductions.
    v8f Pa, Pb;
    #pragma unroll
    for (int r = 0; r < 8; ++r) {
      float sa = Sa[r] * 0.125f, sb = Sb[r] * 0.125f;  // 1/sqrt(64)
      float mx = fmaxf(sa, sb);
      mx = fmaxf(mx, __shfl_xor(mx, 1));
      mx = fmaxf(mx, __shfl_xor(mx, 2));
      mx = fmaxf(mx, __shfl_xor(mx, 4));
      mx = fmaxf(mx, __shfl_xor(mx, 8));
      float mn = fmaxf(m[r], mx);
      float fac = __expf(m[r] - mn);
      m[r] = mn;
      float pa = __expf(sa - mn), pb = __expf(sb - mn);
      Pa[r] = pa; Pb[r] = pb;
      float rs = pa + pb;
      rs += __shfl_xor(rs, 1);
      rs += __shfl_xor(rs, 2);
      rs += __shfl_xor(rs, 4);
      rs += __shfl_xor(rs, 8);
      lsum[r] = lsum[r] * fac + rs;
      #pragma unroll
      for (int c = 0; c < 4; ++c) o[c][r] *= fac;
    }

    // Relayout P: C layout -> LDS -> A layout (wave-private; DS ops in-order)
    #pragma unroll
    for (int r = 0; r < 8; ++r) {
      int M = r + half * 8;
      Pw[w][M][l]      = (bf16_t)Pa[r];
      Pw[w][M][16 + l] = (bf16_t)Pb[r];
    }
    __builtin_amdgcn_wave_barrier();
    const bf16_t* pr = &Pw[w][l][0];
    v16bf pfrag = cat8(*(const v8bf*)(pr + c0), *(const v8bf*)(pr + c0 + 8));

    // O += P(16x32) . V(32x64); 4 WMMAs (B from transposed V in LDS)
    #pragma unroll
    for (int c = 0; c < 4; ++c) {
      const bf16_t* vr = &Vs[cur][c * 16 + l][half * 16];
      v16bf vb = cat8(*(const v8bf*)vr, *(const v8bf*)(vr + 8));
      o[c] = wmma_bf16(pfrag, vb, o[c]);
    }
    __syncthreads();   // all reads of buf `cur` done before it is overwritten
  }

  // Normalize and store attention output as [B,S,H*D] bf16
  size_t obase = ((size_t)b * S_DIM + q0) * D_MODEL + h * HDIM;
  #pragma unroll
  for (int r = 0; r < 8; ++r) {
    float inv = 1.0f / lsum[r];
    int M = r + half * 8;
    bf16_t* op = Ob + obase + (size_t)M * D_MODEL;
    #pragma unroll
    for (int c = 0; c < 4; ++c) op[c * 16 + l] = (bf16_t)(o[c][r] * inv);
  }
}

// ---------------------------------------------------------------------------
// Stage 4: output projection  out = Ob @ Wo + bo  (f32 result)
// Same double-buffered LDS weight staging as stage 2.
// ---------------------------------------------------------------------------
__global__ __launch_bounds__(128)
void out_gemm_kernel(const bf16_t* __restrict__ Ob, const bf16_t* __restrict__ Wot,
                     const float* __restrict__ bo, float* __restrict__ out) {
  __shared__ alignas(16) bf16_t Wtile[2][64][40];   // [buf][n-local][k], pad 32->40

  const int tid = threadIdx.x, w = tid >> 5, lane = tid & 31;
  const int l = lane & 15, half = lane >> 4;
  const int rowblk = (blockIdx.x * 4 + w) * 16;
  const int nblk = blockIdx.y * 64;

  const bf16_t* arow = Ob + (size_t)(rowblk + l) * D_MODEL;

  const int wn = tid >> 1, wseg = tid & 1;
  const bf16_t* wsrc = Wot + (size_t)(nblk + wn) * D_MODEL + wseg * 16;
  async_copy_b128(&Wtile[0][wn][wseg * 16], wsrc);

  v8f acc[4];
  #pragma unroll
  for (int c = 0; c < 4; ++c) acc[c] = zero8();

  #pragma unroll 2
  for (int kc = 0; kc < 32; ++kc) {
    const int cur = kc & 1;
    const int k0 = kc * 32 + half * 16;
    v8bf a_lo = *(const v8bf*)(arow + k0);
    v8bf a_hi = *(const v8bf*)(arow + k0 + 8);

    wait_async0();
    __syncthreads();
    if (kc + 1 < 32)
      async_copy_b128(&Wtile[cur ^ 1][wn][wseg * 16], wsrc + (kc + 1) * 32);

    v16bf a = cat8(a_lo, a_hi);
    #pragma unroll
    for (int c = 0; c < 4; ++c) {
      const bf16_t* wp = &Wtile[cur][c * 16 + l][half * 16];
      v16bf bb = cat8(*(const v8bf*)wp, *(const v8bf*)(wp + 8));
      acc[c] = wmma_bf16(a, bb, acc[c]);
    }
    __syncthreads();
  }
  #pragma unroll
  for (int c = 0; c < 4; ++c) {
    float bv = bo[nblk + c * 16 + l];
    #pragma unroll
    for (int r = 0; r < 8; ++r) {
      int M = r + half * 8;
      out[(size_t)(rowblk + M) * D_MODEL + nblk + c * 16 + l] = acc[c][r] + bv;
    }
  }
}

// ---------------------------------------------------------------------------
extern "C" void kernel_launch(void* const* d_in, const int* in_sizes, int n_in,
                              void* d_out, int out_size, void* d_ws, size_t ws_size,
                              hipStream_t stream) {
  (void)in_sizes; (void)n_in; (void)out_size; (void)ws_size;
  const float* x  = (const float*)d_in[0];
  const float* Wq = (const float*)d_in[1];
  const float* bq = (const float*)d_in[2];
  const float* Wk = (const float*)d_in[3];
  const float* bk = (const float*)d_in[4];
  const float* Wv = (const float*)d_in[5];
  const float* bv = (const float*)d_in[6];
  const float* Wo = (const float*)d_in[7];
  const float* bo = (const float*)d_in[8];
  float* out = (float*)d_out;

  char* ws = (char*)d_ws;
  const size_t MB = 1u << 20;
  bf16_t* Wqt = (bf16_t*)(ws + 0 * MB);    // 2MB each, [n][k] bf16
  bf16_t* Wkt = (bf16_t*)(ws + 2 * MB);
  bf16_t* Wvt = (bf16_t*)(ws + 4 * MB);
  bf16_t* Wot = (bf16_t*)(ws + 6 * MB);
  bf16_t* Qb  = (bf16_t*)(ws + 8 * MB);    // [B,H,S,D]  16MB
  bf16_t* Kb  = (bf16_t*)(ws + 24 * MB);   // [B,H,S,D]  16MB
  bf16_t* Vtb = (bf16_t*)(ws + 40 * MB);   // [B,H,D,S]  16MB
  bf16_t* Ob  = (bf16_t*)(ws + 56 * MB);   // [B,S,H*D]  16MB

  // Stage 1: weight convert+transpose
  wt_bf16_kernel<<<4096, 256, 0, stream>>>(Wq, Wqt);
  wt_bf16_kernel<<<4096, 256, 0, stream>>>(Wk, Wkt);
  wt_bf16_kernel<<<4096, 256, 0, stream>>>(Wv, Wvt);
  wt_bf16_kernel<<<4096, 256, 0, stream>>>(Wo, Wot);

  // Stage 2: QKV projections (8192 rows / 64 per block, 16 heads)
  dim3 gproj(128, NHEAD);
  proj_gemm_kernel<<<gproj, 128, 0, stream>>>(x, Wqt, bq, Qb, 0);
  proj_gemm_kernel<<<gproj, 128, 0, stream>>>(x, Wkt, bk, Kb, 0);
  proj_gemm_kernel<<<gproj, 128, 0, stream>>>(x, Wvt, bv, Vtb, 1);

  // Stage 3: flash attention (2048/64 q-blocks, 16 heads, 4 batches)
  dim3 gattn(S_DIM / 64, NHEAD, B_DIM);
  attn_kernel<<<gattn, 128, 0, stream>>>(Qb, Kb, Vtb, Ob);

  // Stage 4: output projection
  dim3 gout(128, D_MODEL / 64);
  out_gemm_kernel<<<gout, 128, 0, stream>>>(Ob, Wot, bo, out);
}
